// HIST_GAT_77756087927561
// MI455X (gfx1250) — matrix operand order repaired and checked by reading, not compiled, not run on hardware
//
#include <hip/hip_runtime.h>
#include <math.h>

// ---------------- types for WMMA ----------------
typedef __attribute__((ext_vector_type(16))) _Float16 v16h;
typedef __attribute__((ext_vector_type(8)))  _Float16 v8h;
typedef __attribute__((ext_vector_type(8)))  float    v8f;
typedef __attribute__((ext_vector_type(4)))  int      v4i;

#define NROWS 4096
#define HDIM  64
#define GDIM  192      // 3*H
#define TSTEPS 60
#define DDIM  6
#define KPAD0 32       // padded K for layer-0 input GEMM
#define NNELEM ((size_t)NROWS * (size_t)NROWS)

#if __has_builtin(__builtin_amdgcn_global_load_async_to_lds_b128)
#define ASYNC_LDS 1
#else
#define ASYNC_LDS 0
#endif

__device__ __forceinline__ float lkf(float x) { return x >= 0.f ? x : 0.01f * x; }
__device__ __forceinline__ float sigm(float x) { return 1.f / (1.f + __expf(-x)); }

__device__ __forceinline__ void stage16(_Float16* lp, const _Float16* gp) {
#if ASYNC_LDS
  v4i* gpg = (v4i*)(_Float16*)gp;          // drop const, retype (generic)
  v4i* lpg = (v4i*)lp;
  __builtin_amdgcn_global_load_async_to_lds_b128(
      (__attribute__((address_space(1))) v4i*)gpg,
      (__attribute__((address_space(3))) v4i*)lpg, 0, 0);
#else
  uint4 v = *(const uint4*)gp;
  *(uint4*)lp = v;
#endif
}
__device__ __forceinline__ void wait_stage() {
#if ASYNC_LDS
#if __has_builtin(__builtin_amdgcn_s_wait_asynccnt)
  __builtin_amdgcn_s_wait_asynccnt(0);
#else
  asm volatile("s_wait_asynccnt 0" ::: "memory");
#endif
#endif
}

// ---------------- tiny utility kernels ----------------
__global__ void k_zero16(_Float16* p, int n) {
  int i = blockIdx.x * 256 + threadIdx.x;
  if (i < n) p[i] = (_Float16)0.f;
}
__global__ void k_cvt16(_Float16* dst, const float* src, int n) {
  int i = blockIdx.x * 256 + threadIdx.x;
  if (i < n) dst[i] = (_Float16)src[i];
}
// Wih0 (192x6) -> padded f16 (192x32)
__global__ void k_pack_wih0(_Float16* dst, const float* W) {
  int i = blockIdx.x * 256 + threadIdx.x;
  if (i >= GDIM * KPAD0) return;
  int r = i >> 5, c = i & 31;
  dst[i] = (_Float16)(c < DDIM ? W[r * DDIM + c] : 0.f);
}
// per-step layer-0 input: xt16[n][0..31], first 6 cols from x[n, d*T + t]
__global__ void k_pack_xt(_Float16* dst, const float* x, int t) {
  int i = blockIdx.x * 256 + threadIdx.x;
  if (i >= NROWS * KPAD0) return;
  int n = i >> 5, c = i & 31;
  dst[i] = (_Float16)(c < DDIM ? x[(size_t)n * (DDIM * TSTEPS) + c * TSTEPS + t] : 0.f);
}

// ---- WMMA GEMM-NT: C(MxNc,f32) = A(MxK,f16) * B(NcxK,f16)^T --------------
// Block = 8 waves = 128-row M stripe; block-wide 64-col N stripe.
// Each wave computes a 16x64 strip: one A fragment feeds 4 WMMAs per k-step.
// B k-chunks (64 rows x 32 halves = 4KB) are staged in LDS, double-buffered,
// via async global->LDS when available. Requires M%128==0, Nc%64==0, K%32==0.
__global__ __launch_bounds__(256) void k_gemm_nt4(
    float* __restrict__ C, const _Float16* __restrict__ A, int lda,
    const _Float16* __restrict__ B, int ldb,
    int Nc, int K, const float* __restrict__ bias, int act)
{
  __shared__ __align__(32) _Float16 Bs[2][64 * 32];
  int tid = threadIdx.x;
  int wave = tid >> 5, lane = tid & 31;
  int half_id = lane >> 4, l = lane & 15;
  int tm = blockIdx.y * 128 + wave * 16;
  int tn0 = blockIdx.x * 64;

  // cooperative B staging: 256 threads x 16B = 64 rows x 64B chunk
  int br = tid >> 2;            // 0..63 : row within N stripe
  int bseg = (tid & 3) * 8;     // halves within the 32-half chunk
  const _Float16* bsrc = B + (size_t)(tn0 + br) * ldb + bseg;
  int boff = br * 32 + bseg;

  const _Float16* arow = A + (size_t)(tm + l) * lda + half_id * 16;

  v8f acc0 = {}, acc1 = {}, acc2 = {}, acc3 = {};

  stage16(&Bs[0][boff], bsrc);            // chunk 0
  int buf = 0;
  for (int k0 = 0; k0 < K; k0 += 32) {
    wait_stage();
    __syncthreads();                      // Bs[buf] ready; prior reads of buf^1 done
    if (k0 + 32 < K) stage16(&Bs[buf ^ 1][boff], bsrc + k0 + 32);
    union { v16h v; v8h h[2]; } a;
    a.h[0] = *(const v8h*)(arow + k0);
    a.h[1] = *(const v8h*)(arow + k0 + 8);
    const _Float16* bb = &Bs[buf][0];
    v16h b0 = *(const v16h*)(bb + (0 * 16 + l) * 32 + half_id * 16);
    v16h b1 = *(const v16h*)(bb + (1 * 16 + l) * 32 + half_id * 16);
    v16h b2 = *(const v16h*)(bb + (2 * 16 + l) * 32 + half_id * 16);
    v16h b3 = *(const v16h*)(bb + (3 * 16 + l) * 32 + half_id * 16);
    acc0 = __builtin_amdgcn_wmma_f32_16x16x32_f16(false, a.v, false, b0, (short)0, acc0, false, false);
    acc1 = __builtin_amdgcn_wmma_f32_16x16x32_f16(false, a.v, false, b1, (short)0, acc1, false, false);
    acc2 = __builtin_amdgcn_wmma_f32_16x16x32_f16(false, a.v, false, b2, (short)0, acc2, false, false);
    acc3 = __builtin_amdgcn_wmma_f32_16x16x32_f16(false, a.v, false, b3, (short)0, acc3, false, false);
    buf ^= 1;
  }
  int row0 = tm + half_id * 8;
  v8f accs[4] = {acc0, acc1, acc2, acc3};
  for (int q = 0; q < 4; ++q) {
    int col = tn0 + q * 16 + l;
    float bv = bias ? bias[col] : 0.f;
    union { v8f v; float f[8]; } cf; cf.v = accs[q];
    for (int r = 0; r < 8; ++r) {
      float v = cf.f[r] + bv;
      if (act) v = v >= 0.f ? v : 0.01f * v;
      C[(size_t)(row0 + r) * Nc + col] = v;
    }
  }
}

// ---------------- GRU gate fusion ----------------
__global__ void k_gru_gates(_Float16* h16, float* h32,
                            const float* __restrict__ gi, const float* __restrict__ gh,
                            int n_elem)
{
  int idx = blockIdx.x * 256 + threadIdx.x;
  if (idx >= n_elem) return;
  int n = idx >> 6, j = idx & 63;
  const float* gin = gi + (size_t)n * GDIM;
  const float* ghn = gh + (size_t)n * GDIM;
  float r = sigm(gin[j] + ghn[j]);
  float z = sigm(gin[64 + j] + ghn[64 + j]);
  float nn = tanhf(gin[128 + j] + r * ghn[128 + j]);
  float h = (float)h16[idx];
  float hnew = (1.f - z) * nn + z * h;
  h16[idx] = (_Float16)hnew;
  h32[idx] = hnew;
}

// ---------------- attention (rank-1 score) ----------------
__global__ void k_e12(float* e1, float* e2, const float* __restrict__ xt,
                      const float* __restrict__ a)
{
  int n = blockIdx.x * 256 + threadIdx.x;
  if (n >= NROWS) return;
  const float* row = xt + (size_t)n * HDIM;
  float s1 = 0.f, s2 = 0.f;
  for (int j = 0; j < HDIM; ++j) { s1 += row[j] * a[j]; s2 += row[j] * a[HDIM + j]; }
  e1[n] = s1; e2[n] = s2;
}
__global__ void k_max1(float* out, const float* v, int n) {
  __shared__ float red[256];
  int t = threadIdx.x;
  float m = -INFINITY;
  for (int j = t; j < n; j += 256) m = fmaxf(m, v[j]);
  red[t] = m; __syncthreads();
  for (int s = 128; s > 0; s >>= 1) { if (t < s) red[t] = fmaxf(red[t], red[t + s]); __syncthreads(); }
  if (t == 0) out[0] = red[0];
}
__global__ void k_rowstats(float* rowmax, float* rowinv,
                           const float* __restrict__ e1, const float* __restrict__ e2,
                           const float* __restrict__ e1max)
{
  __shared__ float se1[NROWS];
  int t = threadIdx.x;
  for (int j = t; j < NROWS; j += 256) se1[j] = e1[j];
  __syncthreads();
  int i = blockIdx.x * 256 + t;
  float ei = e2[i];
  float m = lkf(ei + e1max[0]);      // leaky is monotone -> row max from global max(e1)
  float s = 0.f;
  for (int j = 0; j < NROWS; ++j) s += __expf(lkf(ei + se1[j]) - m);
  rowmax[i] = m; rowinv[i] = 1.f / s;
}
__global__ void k_att_fill(_Float16* PP, const float* __restrict__ e1,
                           const float* __restrict__ e2,
                           const float* __restrict__ rowmax,
                           const float* __restrict__ rowinv)
{
  size_t idx = (size_t)blockIdx.x * 256 + threadIdx.x;
  int i = (int)(idx >> 12), j = (int)(idx & 4095);
  PP[idx] = (_Float16)(__expf(lkf(e2[i] + e1[j]) - rowmax[i]) * rowinv[i]);
}

// ---------------- elementwise combos ----------------
__global__ void k_add2_cvt(float* dst32, _Float16* dst16, const float* a,
                           const float* b, int n)
{
  int i = blockIdx.x * 256 + threadIdx.x;
  if (i >= n) return;
  float v = a[i] + b[i];
  dst32[i] = v; dst16[i] = (_Float16)v;
}
__global__ void k_sub_cvt(float* dst32, _Float16* dst16, const float* a,
                          const float* b, const float* c, int n)
{
  int i = blockIdx.x * 256 + threadIdx.x;
  if (i >= n) return;
  float v = a[i] - b[i] - (c ? c[i] : 0.f);
  dst32[i] = v; dst16[i] = (_Float16)v;
}

// ---------------- cosine / top-k machinery ----------------
__global__ void k_rownorm(float* norms, const float* __restrict__ A, int M, int K) {
  int m = blockIdx.x * 256 + threadIdx.x;
  if (m >= M) return;
  const float* row = A + (size_t)m * K;
  float q = 0.f;
  for (int k = 0; k < K; ++k) q += row[k] * row[k];
  norms[m] = sqrtf(q);
}
__global__ void k_s_finalize(float* SS, float* diagv, const float* __restrict__ nx) {
  size_t idx = (size_t)blockIdx.x * 256 + threadIdx.x;
  int i = (int)(idx >> 12), j = (int)(idx & 4095);
  float d = nx[i] * nx[j];
  float v = 0.f;
  if (d != 0.f) { v = SS[idx] / d; if (v != v) v = 0.f; }
  if (i == j) { diagv[i] = v; SS[idx] = 0.f; } else SS[idx] = v;
}
__global__ __launch_bounds__(128) void k_topk(float* __restrict__ S) {
  int row = blockIdx.x, t = threadIdx.x;
  float* r = S + (size_t)row * NROWS;
  float v0 = -1e30f, v1 = -1e30f, v2 = -1e30f; int i0 = -1, i1 = -1, i2 = -1;
  for (int j = t; j < NROWS; j += 128) {
    float v = r[j];
    if (v > v0)      { v2 = v1; i2 = i1; v1 = v0; i1 = i0; v0 = v; i0 = j; }
    else if (v > v1) { v2 = v1; i2 = i1; v1 = v; i1 = j; }
    else if (v > v2) { v2 = v; i2 = j; }
  }
  __shared__ float sv[384]; __shared__ int si[384]; __shared__ int g[3];
  sv[t] = v0; sv[128 + t] = v1; sv[256 + t] = v2;
  si[t] = i0; si[128 + t] = i1; si[256 + t] = i2;
  __syncthreads();
  if (t == 0) {
    float b0 = -1e30f, b1 = -1e30f, b2 = -1e30f; int a0 = -1, a1 = -1, a2 = -1;
    for (int q = 0; q < 384; ++q) {
      float v = sv[q]; int id = si[q];
      if (id < 0) continue;
      if (v > b0)      { b2 = b1; a2 = a1; b1 = b0; a1 = a0; b0 = v; a0 = id; }
      else if (v > b1) { b2 = b1; a2 = a1; b1 = v; a1 = id; }
      else if (v > b2) { b2 = v; a2 = id; }
    }
    g[0] = a0; g[1] = a1; g[2] = a2;
  }
  __syncthreads();
  int a0 = g[0], a1 = g[1], a2 = g[2];
  for (int j = t; j < NROWS; j += 128)
    if (j != a0 && j != a1 && j != a2) r[j] = 0.f;
}
__global__ void k_colsum(float* cs, const float* __restrict__ SS) {
  int j = blockIdx.x * 256 + threadIdx.x;
  if (j >= NROWS) return;
  float s = 0.f;
  for (int i = 0; i < NROWS; ++i) s += SS[(size_t)i * NROWS + j];
  cs[j] = s;
}
__global__ void k_adddiag(float* SS, const float* cs, const float* diagv) {
  int i = blockIdx.x * 256 + threadIdx.x;
  if (i >= NROWS) return;
  SS[(size_t)i * NROWS + i] = (cs[i] != 0.f) ? diagv[i] : 0.f;
}
// f16 transpose of NxN f32: dst[j*N+i] = src[i*N+j]
__global__ void k_transNN16(_Float16* dst, const float* __restrict__ src) {
  size_t idx = (size_t)blockIdx.x * 256 + threadIdx.x;
  int j = (int)(idx >> 12), i = (int)(idx & 4095);
  dst[idx] = (_Float16)src[(size_t)i * NROWS + j];
}
// f16 transpose of (N x 64) f32 -> (64 x N): dst[j*N + n] = src[n*64 + j]
__global__ void k_trans64(_Float16* dst, const float* __restrict__ src) {
  size_t idx = (size_t)blockIdx.x * 256 + threadIdx.x;
  int j = (int)(idx >> 12), n = (int)(idx & 4095);
  dst[idx] = (_Float16)src[(size_t)n * HDIM + j];
}
__global__ void k_hidden_post(float* nh, float* keep, _Float16* h16,
                              const float* __restrict__ h32)
{
  int m = blockIdx.x * 256 + threadIdx.x;
  if (m >= NROWS) return;
  const float* row = h32 + (size_t)m * HDIM;
  _Float16* o = h16 + (size_t)m * HDIM;
  float s = 0.f, q = 0.f;
  for (int j = 0; j < HDIM; ++j) {
    float v = row[j]; s += v; q += v * v; o[j] = (_Float16)v;
  }
  keep[m] = (s != 0.f) ? 1.f : 0.f;
  nh[m] = sqrtf(q);
}
__global__ __launch_bounds__(256) void k_wsoftmax(
    _Float16* __restrict__ W, const float* __restrict__ Craw,
    const float* __restrict__ nx, const float* __restrict__ nh,
    const float* __restrict__ keep)
{
  int row = blockIdx.x, t = threadIdx.x;
  __shared__ float buf[NROWS];
  __shared__ float red[256];
  const float* cr = Craw + (size_t)row * NROWS;
  float nxi = nx[row];
  float lmax = -INFINITY;
  for (int j = t; j < NROWS; j += 256) {
    float d = nxi * nh[j];
    float v = 0.f;
    if (d != 0.f) { v = cr[j] / d; if (v != v) v = 0.f; }
    if (keep[j] == 0.f) v = -INFINITY;
    buf[j] = v;
    if (v > lmax) lmax = v;
  }
  red[t] = lmax; __syncthreads();
  for (int s = 128; s > 0; s >>= 1) { if (t < s) red[t] = fmaxf(red[t], red[t + s]); __syncthreads(); }
  float mx = red[0];
  __syncthreads();
  float ls = 0.f;
  for (int j = t; j < NROWS; j += 256) {
    float b = buf[j];
    float e = (b == -INFINITY) ? 0.f : __expf(b - mx);
    buf[j] = e; ls += e;
  }
  red[t] = ls; __syncthreads();
  for (int s = 128; s > 0; s >>= 1) { if (t < s) red[t] += red[t + s]; __syncthreads(); }
  float inv = red[0] > 0.f ? 1.f / red[0] : 0.f;
  _Float16* wr = W + (size_t)row * NROWS;
  for (int j = t; j < NROWS; j += 256) wr[j] = (_Float16)(buf[j] * inv);
}
__global__ void k_final(float* out, const float* og, const float* oh, const float* oi,
                        const float* __restrict__ Wo, const float* __restrict__ bo)
{
  int n = blockIdx.x * 256 + threadIdx.x;
  if (n >= NROWS) return;
  size_t base = (size_t)n * HDIM;
  float s = 0.f;
  for (int j = 0; j < HDIM; ++j)
    s += (og[base + j] + oh[base + j] + oi[base + j]) * Wo[j];
  out[n] = s + bo[0];
}

// ---------------- host side ----------------
#define CDIV(a, b) (((a) + (b) - 1) / (b))

static void gemm(hipStream_t st, float* C, const _Float16* A, int lda,
                 const _Float16* B, int ldb, int M, int Nc, int K,
                 const float* bias, int act)
{
  dim3 grid(Nc / 64, M / 128);
  k_gemm_nt4<<<grid, 256, 0, st>>>(C, A, lda, B, ldb, Nc, K, bias, act);
}

extern "C" void kernel_launch(void* const* d_in, const int* in_sizes, int n_in,
                              void* d_out, int out_size, void* d_ws, size_t ws_size,
                              hipStream_t stream)
{
  (void)in_sizes; (void)n_in; (void)out_size; (void)ws_size;
  const float* x    = (const float*)d_in[0];
  const float* Wih0 = (const float*)d_in[1];
  const float* Whh0 = (const float*)d_in[2];
  const float* bih0 = (const float*)d_in[3];
  const float* bhh0 = (const float*)d_in[4];
  const float* Wih1 = (const float*)d_in[5];
  const float* Whh1 = (const float*)d_in[6];
  const float* bih1 = (const float*)d_in[7];
  const float* bhh1 = (const float*)d_in[8];
  const float* Wt   = (const float*)d_in[9];
  const float* bt   = (const float*)d_in[10];
  const float* av   = (const float*)d_in[11];
  const float* Wg   = (const float*)d_in[12];
  const float* bg   = (const float*)d_in[13];
  const float* Wgf  = (const float*)d_in[14];
  const float* bgf  = (const float*)d_in[15];
  const float* Wgb  = (const float*)d_in[16];
  const float* bgb  = (const float*)d_in[17];
  const float* Whs  = (const float*)d_in[18];
  const float* bhs  = (const float*)d_in[19];
  const float* Whsf = (const float*)d_in[20];
  const float* bhsf = (const float*)d_in[21];
  const float* Whsb = (const float*)d_in[22];
  const float* bhsb = (const float*)d_in[23];
  const float* Wi   = (const float*)d_in[24];
  const float* bi   = (const float*)d_in[25];
  const float* Wo   = (const float*)d_in[26];
  const float* bo   = (const float*)d_in[27];
  float* out = (float*)d_out;

  // ---- bump allocator over d_ws ----
  char* base = (char*)d_ws;
  size_t off = 0;
  auto alloc = [&](size_t bytes) -> void* {
    void* p = base + off;
    off = (off + bytes + 255) & ~(size_t)255;
    return p;
  };
  const size_t NH = (size_t)NROWS * HDIM;
  float*    SS   = (float*)alloc(NNELEM * 4);
  _Float16* PP   = (_Float16*)alloc(NNELEM * 2);
  float*    gi   = (float*)alloc((size_t)NROWS * GDIM * 4);
  float*    gh   = (float*)alloc((size_t)NROWS * GDIM * 4);
  _Float16* w16_ih0p = (_Float16*)alloc(GDIM * KPAD0 * 2);
  _Float16* w16_hh0  = (_Float16*)alloc(GDIM * HDIM * 2);
  _Float16* w16_ih1  = (_Float16*)alloc(GDIM * HDIM * 2);
  _Float16* w16_hh1  = (_Float16*)alloc(GDIM * HDIM * 2);
  _Float16* w16_t    = (_Float16*)alloc(HDIM * HDIM * 2);
  _Float16* w16_g    = (_Float16*)alloc(HDIM * HDIM * 2);
  _Float16* w16_gf   = (_Float16*)alloc(HDIM * HDIM * 2);
  _Float16* w16_gb   = (_Float16*)alloc(HDIM * HDIM * 2);
  _Float16* w16_hs   = (_Float16*)alloc(HDIM * HDIM * 2);
  _Float16* w16_hsf  = (_Float16*)alloc(HDIM * HDIM * 2);
  _Float16* w16_hsb  = (_Float16*)alloc(HDIM * HDIM * 2);
  _Float16* w16_i    = (_Float16*)alloc(HDIM * HDIM * 2);
  _Float16* xt16     = (_Float16*)alloc((size_t)NROWS * KPAD0 * 2);
  _Float16* h1_16    = (_Float16*)alloc(NH * 2);
  _Float16* h2_16    = (_Float16*)alloc(NH * 2);
  _Float16* XhT16    = (_Float16*)alloc(NH * 2);
  _Float16* hsi16    = (_Float16*)alloc(NH * 2);
  _Float16* hsiT16   = (_Float16*)alloc(NH * 2);
  _Float16* hid16    = (_Float16*)alloc(NH * 2);
  _Float16* hidT16   = (_Float16*)alloc(NH * 2);
  _Float16* gat16    = (_Float16*)alloc(NH * 2);
  _Float16* u16      = (_Float16*)alloc(NH * 2);
  _Float16* wh16     = (_Float16*)alloc(NH * 2);
  _Float16* hsh16    = (_Float16*)alloc(NH * 2);
  _Float16* indi16   = (_Float16*)alloc(NH * 2);
  float* h1_32   = (float*)alloc(NH * 4);
  float* h2_32   = (float*)alloc(NH * 4);
  float* xt32    = (float*)alloc(NH * 4);
  float* attXh   = (float*)alloc(NH * 4);
  float* u32     = (float*)alloc(NH * 4);
  float* gat32   = (float*)alloc(NH * 4);
  float* gcnb    = (float*)alloc(NH * 4);
  float* ogcn    = (float*)alloc(NH * 4);
  float* hsi32   = (float*)alloc(NH * 4);
  float* hid32   = (float*)alloc(NH * 4);
  float* wh32    = (float*)alloc(NH * 4);
  float* hsh32   = (float*)alloc(NH * 4);
  float* hsback  = (float*)alloc(NH * 4);
  float* ohs     = (float*)alloc(NH * 4);
  float* oindi   = (float*)alloc(NH * 4);
  float* indi32  = (float*)alloc(NH * 4);
  float* e1      = (float*)alloc(NROWS * 4);
  float* e2      = (float*)alloc(NROWS * 4);
  float* e1max   = (float*)alloc(256);
  float* rowmax  = (float*)alloc(NROWS * 4);
  float* rowinv  = (float*)alloc(NROWS * 4);
  float* nx      = (float*)alloc(NROWS * 4);
  float* diagv   = (float*)alloc(NROWS * 4);
  float* cs      = (float*)alloc(NROWS * 4);
  float* nh      = (float*)alloc(NROWS * 4);
  float* keep    = (float*)alloc(NROWS * 4);

  const int EB = 256;
  const int NNB = (int)(NNELEM / EB);

  // ---- weight conversion ----
  k_pack_wih0<<<CDIV(GDIM * KPAD0, EB), EB, 0, stream>>>(w16_ih0p, Wih0);
  k_cvt16<<<CDIV(GDIM * HDIM, EB), EB, 0, stream>>>(w16_hh0, Whh0, GDIM * HDIM);
  k_cvt16<<<CDIV(GDIM * HDIM, EB), EB, 0, stream>>>(w16_ih1, Wih1, GDIM * HDIM);
  k_cvt16<<<CDIV(GDIM * HDIM, EB), EB, 0, stream>>>(w16_hh1, Whh1, GDIM * HDIM);
  k_cvt16<<<CDIV(HDIM * HDIM, EB), EB, 0, stream>>>(w16_t, Wt, HDIM * HDIM);
  k_cvt16<<<CDIV(HDIM * HDIM, EB), EB, 0, stream>>>(w16_g, Wg, HDIM * HDIM);
  k_cvt16<<<CDIV(HDIM * HDIM, EB), EB, 0, stream>>>(w16_gf, Wgf, HDIM * HDIM);
  k_cvt16<<<CDIV(HDIM * HDIM, EB), EB, 0, stream>>>(w16_gb, Wgb, HDIM * HDIM);
  k_cvt16<<<CDIV(HDIM * HDIM, EB), EB, 0, stream>>>(w16_hs, Whs, HDIM * HDIM);
  k_cvt16<<<CDIV(HDIM * HDIM, EB), EB, 0, stream>>>(w16_hsf, Whsf, HDIM * HDIM);
  k_cvt16<<<CDIV(HDIM * HDIM, EB), EB, 0, stream>>>(w16_hsb, Whsb, HDIM * HDIM);
  k_cvt16<<<CDIV(HDIM * HDIM, EB), EB, 0, stream>>>(w16_i, Wi, HDIM * HDIM);
  k_zero16<<<CDIV((int)NH, EB), EB, 0, stream>>>(h1_16, (int)NH);
  k_zero16<<<CDIV((int)NH, EB), EB, 0, stream>>>(h2_16, (int)NH);

  // ---- GRU: 60 steps x 2 layers, all matmuls via WMMA GEMM-NT ----
  for (int t = 0; t < TSTEPS; ++t) {
    k_pack_xt<<<CDIV(NROWS * KPAD0, EB), EB, 0, stream>>>(xt16, x, t);
    gemm(stream, gi, xt16, KPAD0, w16_ih0p, KPAD0, NROWS, GDIM, KPAD0, bih0, 0);
    gemm(stream, gh, h1_16, HDIM, w16_hh0, HDIM, NROWS, GDIM, HDIM, bhh0, 0);
    k_gru_gates<<<CDIV((int)NH, EB), EB, 0, stream>>>(h1_16, h1_32, gi, gh, (int)NH);
    gemm(stream, gi, h1_16, HDIM, w16_ih1, HDIM, NROWS, GDIM, HDIM, bih1, 0);
    gemm(stream, gh, h2_16, HDIM, w16_hh1, HDIM, NROWS, GDIM, HDIM, bhh1, 0);
    k_gru_gates<<<CDIV((int)NH, EB), EB, 0, stream>>>(h2_16, h2_32, gi, gh, (int)NH);
  }

  // ---- attention: xt = Xh@Wt^T + bt ; rank-1 score softmax ----
  gemm(stream, xt32, h2_16, HDIM, w16_t, HDIM, NROWS, HDIM, HDIM, bt, 0);
  k_e12<<<CDIV(NROWS, EB), EB, 0, stream>>>(e1, e2, xt32, av);
  k_max1<<<1, 256, 0, stream>>>(e1max, e1, NROWS);
  k_rowstats<<<NROWS / 256, 256, 0, stream>>>(rowmax, rowinv, e1, e2, e1max);
  k_att_fill<<<NNB, EB, 0, stream>>>(PP, e1, e2, rowmax, rowinv);
  k_trans64<<<CDIV((int)NH, EB), EB, 0, stream>>>(XhT16, h2_32);
  gemm(stream, attXh, PP, NROWS, XhT16, NROWS, NROWS, HDIM, NROWS, nullptr, 0);

  // ---- GCN stage ----
  k_add2_cvt<<<CDIV((int)NH, EB), EB, 0, stream>>>(u32, u16, attXh, h2_32, (int)NH);
  gemm(stream, gat32, u16, HDIM, w16_g, HDIM, NROWS, HDIM, HDIM, bg, 1);
  k_cvt16<<<CDIV((int)NH, EB), EB, 0, stream>>>(gat16, gat32, (int)NH);
  gemm(stream, gcnb, gat16, HDIM, w16_gb, HDIM, NROWS, HDIM, HDIM, bgb, 0);
  gemm(stream, ogcn, gat16, HDIM, w16_gf, HDIM, NROWS, HDIM, HDIM, bgf, 1);
  k_sub_cvt<<<CDIV((int)NH, EB), EB, 0, stream>>>(hsi32, hsi16, h2_32, gcnb, nullptr, (int)NH);

  // ---- cosine graph + top-k ----
  k_rownorm<<<CDIV(NROWS, EB), EB, 0, stream>>>(nx, hsi32, NROWS, HDIM);
  gemm(stream, SS, hsi16, HDIM, hsi16, HDIM, NROWS, NROWS, HDIM, nullptr, 0);
  k_s_finalize<<<NNB, EB, 0, stream>>>(SS, diagv, nx);
  k_topk<<<NROWS, 128, 0, stream>>>(SS);
  k_colsum<<<CDIV(NROWS, EB), EB, 0, stream>>>(cs, SS);
  k_adddiag<<<CDIV(NROWS, EB), EB, 0, stream>>>(SS, cs, diagv);
  k_transNN16<<<NNB, EB, 0, stream>>>(PP, SS);            // PP = s^T (f16)
  k_trans64<<<CDIV((int)NH, EB), EB, 0, stream>>>(hsiT16, hsi32);
  gemm(stream, hid32, PP, NROWS, hsiT16, NROWS, NROWS, HDIM, NROWS, nullptr, 0);

  // ---- shared-hidden stage ----
  k_hidden_post<<<CDIV(NROWS, EB), EB, 0, stream>>>(nh, keep, hid16, hid32);
  k_trans64<<<CDIV((int)NH, EB), EB, 0, stream>>>(hidT16, hid32);
  gemm(stream, SS, hsi16, HDIM, hid16, HDIM, NROWS, NROWS, HDIM, nullptr, 0); // c raw
  k_wsoftmax<<<NROWS, 256, 0, stream>>>(PP, SS, nx, nh, keep);                 // PP = w
  gemm(stream, wh32, PP, NROWS, hidT16, NROWS, NROWS, HDIM, NROWS, nullptr, 0);
  k_cvt16<<<CDIV((int)NH, EB), EB, 0, stream>>>(wh16, wh32, (int)NH);
  gemm(stream, hsh32, wh16, HDIM, w16_hs, HDIM, NROWS, HDIM, HDIM, bhs, 0);
  k_cvt16<<<CDIV((int)NH, EB), EB, 0, stream>>>(hsh16, hsh32, (int)NH);
  gemm(stream, hsback, hsh16, HDIM, w16_hsb, HDIM, NROWS, HDIM, HDIM, bhsb, 0);
  gemm(stream, ohs, hsh16, HDIM, w16_hsf, HDIM, NROWS, HDIM, HDIM, bhsf, 1);

  // ---- individual stage + output ----
  k_sub_cvt<<<CDIV((int)NH, EB), EB, 0, stream>>>(indi32, indi16, h2_32, gcnb, hsback, (int)NH);
  gemm(stream, oindi, indi16, HDIM, w16_i, HDIM, NROWS, HDIM, HDIM, bi, 1);
  k_final<<<CDIV(NROWS, EB), EB, 0, stream>>>(out, ogcn, ohs, oindi, Wo, bo);
}
